// RSSMCell_25598005084986
// MI455X (gfx1250) — matrix operand order repaired and checked by reading, not compiled, not run on hardware
//
#include <hip/hip_runtime.h>
#include <hip/hip_bf16.h>

// Problem dims
#define BB   16384
#define HH   1024
#define LL   512
#define AACT 12
#define KSA  524
#define KSAP 544   // 524 padded to multiple of 32

typedef __attribute__((ext_vector_type(16))) __bf16 bf16x16;
typedef __attribute__((ext_vector_type(8)))  float  f32x8;

__device__ __forceinline__ float sigmoidf_(float x) { return 1.0f / (1.0f + expf(-x)); }

__device__ __forceinline__ f32x8 wmma_bf16(bf16x16 a, bf16x16 b, f32x8 c) {
    return __builtin_amdgcn_wmma_f32_16x16x32_bf16(false, a, false, b, (short)0, c, false, false);
}

// A fragment (16x32 bf16, row-major source). Per CDNA5 ISA layout:
// lane L holds row M = L&15; K-halves [g*8, g*8+8) and [16+g*8, 24+g*8) with g = L>>4.
__device__ __forceinline__ bf16x16 load_a_frag(const __bf16* __restrict__ X, int lda,
                                               int row_base, int k0, int lane) {
    const __bf16* p = X + (size_t)(row_base + (lane & 15)) * lda + (k0 + ((lane >> 4) << 3));
    union { uint4 u[2]; bf16x16 v; } t;
    t.u[0] = *(const uint4*)(p);
    t.u[1] = *(const uint4*)(p + 16);
    return t.v;
}

// B fragment (32x16 bf16) from W^T stored row-major [N, ldk]:
// lane L holds column N = n0 + (L&15); K-halves [k0+g*8, +8) and [k0+16+g*8, +8).
__device__ __forceinline__ bf16x16 load_b_frag(const __bf16* __restrict__ WT, int ldk,
                                               int n0, int k0, int lane) {
    const __bf16* p = WT + (size_t)(n0 + (lane & 15)) * ldk + (k0 + ((lane >> 4) << 3));
    union { uint4 u[2]; bf16x16 v; } t;
    t.u[0] = *(const uint4*)(p);
    t.u[1] = *(const uint4*)(p + 16);
    return t.v;
}

// ---------------------------------------------------------------------------
// Fused Linear (+bias) [+ LayerNorm + ELU -> bf16]  or  [plain -> f32]
// Workgroup: 32 rows x 1024 cols. 8 waves; each wave: 8 column tiles x 2 row
// sub-tiles (16 WMMAs per K-step sharing 8 B-fragments + 2 A-fragments).
// EPI==0: y = ELU(LN(XW+b)) -> bf16 ; EPI==1: y = XW+b -> f32
// ---------------------------------------------------------------------------
template <int EPI>
__global__ __launch_bounds__(256)
void fused_linear_kernel(const __bf16* __restrict__ X, int lda, int Kp,
                         const __bf16* __restrict__ WT,
                         const float* __restrict__ bias,
                         const float* __restrict__ gamma,
                         const float* __restrict__ beta,
                         float eps,
                         __bf16* __restrict__ Ybf, int ldy,
                         float* __restrict__ Yf) {
    __shared__ float red_s[32];
    __shared__ float red_q[32];

    const int lane = threadIdx.x & 31;
    const int w    = threadIdx.x >> 5;
    const int g    = lane >> 4;
    const int cl   = lane & 15;
    const int row_base = blockIdx.x * 32;

    if (EPI == 0) {
        if (threadIdx.x < 32) { red_s[threadIdx.x] = 0.0f; red_q[threadIdx.x] = 0.0f; }
        __syncthreads();
    }

    f32x8 acc[2][8];
#pragma unroll
    for (int rt = 0; rt < 2; ++rt)
#pragma unroll
        for (int t = 0; t < 8; ++t) acc[rt][t] = (f32x8)(0.0f);

    for (int k0 = 0; k0 < Kp; k0 += 32) {
        bf16x16 a0 = load_a_frag(X, lda, row_base,      k0, lane);
        bf16x16 a1 = load_a_frag(X, lda, row_base + 16, k0, lane);
#pragma unroll
        for (int t = 0; t < 8; ++t) {
            bf16x16 b = load_b_frag(WT, Kp, w * 128 + t * 16, k0, lane);
            acc[0][t] = wmma_bf16(a0, b, acc[0][t]);
            acc[1][t] = wmma_bf16(a1, b, acc[1][t]);
        }
    }

    // bias + LN partial stats
    float s[2][8], q[2][8];
#pragma unroll
    for (int rt = 0; rt < 2; ++rt)
#pragma unroll
        for (int i = 0; i < 8; ++i) { s[rt][i] = 0.0f; q[rt][i] = 0.0f; }

#pragma unroll
    for (int t = 0; t < 8; ++t) {
        const int col = w * 128 + t * 16 + cl;
        const float bc = bias[col];
#pragma unroll
        for (int rt = 0; rt < 2; ++rt) {
#pragma unroll
            for (int i = 0; i < 8; ++i) {
                float v = acc[rt][t][i] + bc;
                acc[rt][t][i] = v;
                if (EPI == 0) { s[rt][i] += v; q[rt][i] += v * v; }
            }
        }
    }

    if (EPI == 0) {
#pragma unroll
        for (int rt = 0; rt < 2; ++rt) {
#pragma unroll
            for (int i = 0; i < 8; ++i) {
#pragma unroll
                for (int m = 1; m < 16; m <<= 1) {
                    s[rt][i] += __shfl_xor(s[rt][i], m, 16);
                    q[rt][i] += __shfl_xor(q[rt][i], m, 16);
                }
            }
        }
        if (cl == 0) {
#pragma unroll
            for (int rt = 0; rt < 2; ++rt)
#pragma unroll
                for (int i = 0; i < 8; ++i) {
                    atomicAdd(&red_s[rt * 16 + i + 8 * g], s[rt][i]);
                    atomicAdd(&red_q[rt * 16 + i + 8 * g], q[rt][i]);
                }
        }
        __syncthreads();
    }

#pragma unroll
    for (int rt = 0; rt < 2; ++rt) {
#pragma unroll
        for (int i = 0; i < 8; ++i) {
            const int rl = rt * 16 + i + 8 * g;
            const size_t row = (size_t)(row_base + rl);
            float mean = 0.0f, inv = 1.0f;
            if (EPI == 0) {
                mean = red_s[rl] * (1.0f / 1024.0f);
                float var = red_q[rl] * (1.0f / 1024.0f) - mean * mean;
                inv = rsqrtf(var + eps);
            }
#pragma unroll
            for (int t = 0; t < 8; ++t) {
                const int col = w * 128 + t * 16 + cl;
                float v = acc[rt][t][i];
                if (EPI == 0) {
                    float y = (v - mean) * inv * gamma[col] + beta[col];
                    y = y > 0.0f ? y : expm1f(y);
                    Ybf[row * (size_t)ldy + col] = (__bf16)y;
                } else {
                    Yf[row * 1024 + col] = v;
                }
            }
        }
    }
}

// ---------------------------------------------------------------------------
// GRU: gi = sa@Wi+bi, gh = h@Wh+bh, gates -> belief_new (f32 out + bf16 into Xpost)
// Workgroup: 32 rows x 128 cols; each wave one 16-col tile x 2 row sub-tiles.
// 12 WMMAs per K-step sharing 6 B-fragments + 4 A-fragments.
// ---------------------------------------------------------------------------
__global__ __launch_bounds__(256)
void gru_kernel(const __bf16* __restrict__ sa, const __bf16* __restrict__ hb,
                const __bf16* __restrict__ WiT, const __bf16* __restrict__ WhT,
                const float* __restrict__ bi, const float* __restrict__ bh,
                const float* __restrict__ hf,
                float* __restrict__ bn, __bf16* __restrict__ xpost) {
    const int lane = threadIdx.x & 31;
    const int w    = threadIdx.x >> 5;
    const int g    = lane >> 4;
    const int cl   = lane & 15;
    const int row_base = blockIdx.x * 32;
    const int n0 = blockIdx.y * 128 + w * 16;

    f32x8 gi[3][2], gh[3][2];
#pragma unroll
    for (int e = 0; e < 3; ++e)
#pragma unroll
        for (int rt = 0; rt < 2; ++rt) { gi[e][rt] = (f32x8)(0.0f); gh[e][rt] = (f32x8)(0.0f); }

    for (int k0 = 0; k0 < HH; k0 += 32) {
        bf16x16 as0 = load_a_frag(sa, HH, row_base,      k0, lane);
        bf16x16 as1 = load_a_frag(sa, HH, row_base + 16, k0, lane);
        bf16x16 ah0 = load_a_frag(hb, HH, row_base,      k0, lane);
        bf16x16 ah1 = load_a_frag(hb, HH, row_base + 16, k0, lane);
#pragma unroll
        for (int e = 0; e < 3; ++e) {
            bf16x16 bwi = load_b_frag(WiT, HH, n0 + e * 1024, k0, lane);
            gi[e][0] = wmma_bf16(as0, bwi, gi[e][0]);
            gi[e][1] = wmma_bf16(as1, bwi, gi[e][1]);
            bf16x16 bwh = load_b_frag(WhT, HH, n0 + e * 1024, k0, lane);
            gh[e][0] = wmma_bf16(ah0, bwh, gh[e][0]);
            gh[e][1] = wmma_bf16(ah1, bwh, gh[e][1]);
        }
    }

    const int col = n0 + cl;
    const float bir = bi[col], biz = bi[col + 1024], bin = bi[col + 2048];
    const float bhr = bh[col], bhz = bh[col + 1024], bhn = bh[col + 2048];
#pragma unroll
    for (int rt = 0; rt < 2; ++rt) {
#pragma unroll
        for (int i = 0; i < 8; ++i) {
            const size_t row = (size_t)(row_base + rt * 16 + i + 8 * g);
            float ir  = gi[0][rt][i] + bir;
            float iz  = gi[1][rt][i] + biz;
            float inn = gi[2][rt][i] + bin;
            float hr  = gh[0][rt][i] + bhr;
            float hz  = gh[1][rt][i] + bhz;
            float hn  = gh[2][rt][i] + bhn;
            float r = sigmoidf_(ir + hr);
            float z = sigmoidf_(iz + hz);
            float n = tanhf(inn + r * hn);
            float h = hf[row * 1024 + col];
            float o = (1.0f - z) * n + z * h;
            bn[row * 1024 + col] = o;
            xpost[row * 2048 + col] = (__bf16)o;
        }
    }
}

// ---------------------------------------------------------------------------
// Elementwise helpers
// ---------------------------------------------------------------------------
__global__ void sample_kernel(const float* __restrict__ pp, const float* __restrict__ eps,
                              float* __restrict__ out, int n) {
    int idx = blockIdx.x * blockDim.x + threadIdx.x;
    if (idx >= n) return;
    int row = idx >> 9;          // /512
    int j   = idx & 511;
    float mean = pp[(size_t)row * 1024 + j];
    float sr   = pp[(size_t)row * 1024 + 512 + j];
    float std  = 2.0f * sigmoidf_(sr) + 0.1f;
    out[idx] = mean + std * eps[idx];
}

__global__ void cast_copy_kernel(const float* __restrict__ src, __bf16* __restrict__ dst,
                                 long n, int cols, int dst_ld, int col_off) {
    long idx = (long)blockIdx.x * blockDim.x + threadIdx.x;
    if (idx >= n) return;
    long r = idx / cols;
    int  c = (int)(idx - r * cols);
    dst[(size_t)r * dst_ld + col_off + c] = (__bf16)src[idx];
}

__global__ void build_xsa_kernel(const float* __restrict__ latent, const float* __restrict__ action,
                                 __bf16* __restrict__ dst, long n) {
    long idx = (long)blockIdx.x * blockDim.x + threadIdx.x;
    if (idx >= n) return;
    long r = idx / KSAP;
    int  c = (int)(idx - r * KSAP);
    float v = (c < LL) ? latent[r * LL + c]
            : (c < KSA) ? action[r * AACT + (c - LL)]
            : 0.0f;
    dst[idx] = (__bf16)v;
}

// W [K,N] f32 (row-major) -> W^T bf16 [N, Kp], zero-padded K
__global__ void transpose_cast_kernel(const float* __restrict__ src, __bf16* __restrict__ dst,
                                      int K, int N, int Kp) {
    long idx = (long)blockIdx.x * blockDim.x + threadIdx.x;
    long n = (long)N * Kp;
    if (idx >= n) return;
    int nn = (int)(idx / Kp);
    int k  = (int)(idx - (long)nn * Kp);
    dst[idx] = (__bf16)((k < K) ? src[(size_t)k * N + nn] : 0.0f);
}

// ---------------------------------------------------------------------------
extern "C" void kernel_launch(void* const* d_in, const int* in_sizes, int n_in,
                              void* d_out, int out_size, void* d_ws, size_t ws_size,
                              hipStream_t stream) {
    (void)in_sizes; (void)n_in; (void)out_size; (void)ws_size;

    const float* belief = (const float*)d_in[0];
    const float* latent = (const float*)d_in[1];
    const float* action = (const float*)d_in[2];
    const float* obs    = (const float*)d_in[3];
    const float* eps_pr = (const float*)d_in[4];
    const float* eps_po = (const float*)d_in[5];
    const float* W_sa   = (const float*)d_in[6];
    const float* b_sa   = (const float*)d_in[7];
    const float* g_sa   = (const float*)d_in[8];
    const float* be_sa  = (const float*)d_in[9];
    const float* Wi     = (const float*)d_in[10];
    const float* bi     = (const float*)d_in[11];
    const float* Wh     = (const float*)d_in[12];
    const float* bh     = (const float*)d_in[13];
    // prior_mlp: W1 b1 g1 be1 W2 b2 g2 be2 W3 b3  (indices 14..23)
    const float* pW1 = (const float*)d_in[14]; const float* pb1 = (const float*)d_in[15];
    const float* pg1 = (const float*)d_in[16]; const float* pbe1= (const float*)d_in[17];
    const float* pW2 = (const float*)d_in[18]; const float* pb2 = (const float*)d_in[19];
    const float* pg2 = (const float*)d_in[20]; const float* pbe2= (const float*)d_in[21];
    const float* pW3 = (const float*)d_in[22]; const float* pb3 = (const float*)d_in[23];
    // post_mlp (indices 24..33)
    const float* qW1 = (const float*)d_in[24]; const float* qb1 = (const float*)d_in[25];
    const float* qg1 = (const float*)d_in[26]; const float* qbe1= (const float*)d_in[27];
    const float* qW2 = (const float*)d_in[28]; const float* qb2 = (const float*)d_in[29];
    const float* qg2 = (const float*)d_in[30]; const float* qbe2= (const float*)d_in[31];
    const float* qW3 = (const float*)d_in[32]; const float* qb3 = (const float*)d_in[33];

    float* out = (float*)d_out;
    float* o_prior = out;                              // [B,1024]
    float* o_ps    = out + (size_t)BB * 1024;          // [B,512]
    float* o_post  = out + (size_t)BB * 1536;          // [B,1024]
    float* o_qs    = out + (size_t)BB * 2560;          // [B,512]
    float* o_bn    = out + (size_t)BB * 3072;          // [B,1024]

    __bf16* ws = (__bf16*)d_ws;
    size_t off = 0;
    auto alloc = [&](size_t nelem) { size_t o = off; off += (nelem + 127) & ~(size_t)127; return o; };
    __bf16* Xsa   = ws + alloc((size_t)BB * KSAP);
    __bf16* SAb   = ws + alloc((size_t)BB * HH);
    __bf16* Hb    = ws + alloc((size_t)BB * HH);
    __bf16* Xpost = ws + alloc((size_t)BB * 2048);
    __bf16* H1    = ws + alloc((size_t)BB * HH);
    __bf16* H2    = ws + alloc((size_t)BB * HH);
    __bf16* WsaT  = ws + alloc((size_t)HH * KSAP);
    __bf16* WiT   = ws + alloc((size_t)3 * HH * HH);
    __bf16* WhT   = ws + alloc((size_t)3 * HH * HH);
    __bf16* PW1T  = ws + alloc((size_t)HH * HH);
    __bf16* PW2T  = ws + alloc((size_t)HH * HH);
    __bf16* PW3T  = ws + alloc((size_t)HH * HH);
    __bf16* QW1T  = ws + alloc((size_t)HH * 2048);
    __bf16* QW2T  = ws + alloc((size_t)HH * HH);
    __bf16* QW3T  = ws + alloc((size_t)HH * HH);

    auto tr = [&](const float* W, __bf16* WT, int K, int N, int Kp) {
        long n = (long)N * Kp;
        transpose_cast_kernel<<<(unsigned)((n + 255) / 256), 256, 0, stream>>>(W, WT, K, N, Kp);
    };

    // --- weight prep: fp32 [K,N] -> bf16 W^T [N,Kp] (stays hot in 192MB L2) ---
    tr(W_sa, WsaT, KSA, HH, KSAP);
    tr(Wi,   WiT,  HH, 3 * HH, HH);
    tr(Wh,   WhT,  HH, 3 * HH, HH);
    tr(pW1, PW1T, HH, HH, HH);
    tr(pW2, PW2T, HH, HH, HH);
    tr(pW3, PW3T, HH, HH, HH);
    tr(qW1, QW1T, 2048, HH, 2048);
    tr(qW2, QW2T, HH, HH, HH);
    tr(qW3, QW3T, HH, HH, HH);

    // --- activation prep ---
    {
        long n = (long)BB * KSAP;
        build_xsa_kernel<<<(unsigned)((n + 255) / 256), 256, 0, stream>>>(latent, action, Xsa, n);
    }
    {
        long n = (long)BB * HH;
        cast_copy_kernel<<<(unsigned)((n + 255) / 256), 256, 0, stream>>>(belief, Hb, n, HH, HH, 0);
        cast_copy_kernel<<<(unsigned)((n + 255) / 256), 256, 0, stream>>>(obs, Xpost, n, HH, 2048, 1024);
    }

    const unsigned GR = BB / 32;  // 512 row-tiles of 32 rows

    // sa = ELU(LN(concat(ls,act)@W_sa + b, eps=1e-3))
    fused_linear_kernel<0><<<GR, 256, 0, stream>>>(Xsa, KSAP, KSAP, WsaT, b_sa, g_sa, be_sa,
                                                   1e-3f, SAb, HH, nullptr);
    // GRU -> belief_new (f32 out + bf16 into Xpost[:, :1024])
    gru_kernel<<<dim3(GR, 8), 256, 0, stream>>>(SAb, Hb, WiT, WhT, bi, bh, belief, o_bn, Xpost);

    // prior MLP (reads Xpost cols 0..1023, lda=2048)
    fused_linear_kernel<0><<<GR, 256, 0, stream>>>(Xpost, 2048, HH, PW1T, pb1, pg1, pbe1,
                                                   1e-5f, H1, HH, nullptr);
    fused_linear_kernel<0><<<GR, 256, 0, stream>>>(H1, HH, HH, PW2T, pb2, pg2, pbe2,
                                                   1e-5f, H2, HH, nullptr);
    fused_linear_kernel<1><<<GR, 256, 0, stream>>>(H2, HH, HH, PW3T, pb3, nullptr, nullptr,
                                                   0.0f, nullptr, 0, o_prior);
    sample_kernel<<<(BB * 512 + 255) / 256, 256, 0, stream>>>(o_prior, eps_pr, o_ps, BB * 512);

    // posterior MLP (reads full Xpost [B,2048])
    fused_linear_kernel<0><<<GR, 256, 0, stream>>>(Xpost, 2048, 2048, QW1T, qb1, qg1, qbe1,
                                                   1e-5f, H1, HH, nullptr);
    fused_linear_kernel<0><<<GR, 256, 0, stream>>>(H1, HH, HH, QW2T, qb2, qg2, qbe2,
                                                   1e-5f, H2, HH, nullptr);
    fused_linear_kernel<1><<<GR, 256, 0, stream>>>(H2, HH, HH, QW3T, qb3, nullptr, nullptr,
                                                   0.0f, nullptr, 0, o_post);
    sample_kernel<<<(BB * 512 + 255) / 256, 256, 0, stream>>>(o_post, eps_po, o_qs, BB * 512);
}
